// DefaultSegmentLinear_30150670418080
// MI455X (gfx1250) — matrix operand order repaired and checked by reading, not compiled, not run on hardware
//
#include <hip/hip_runtime.h>

typedef __attribute__((ext_vector_type(16))) __bf16 v16bf;
typedef __attribute__((ext_vector_type(8)))  float  v8f;

#define BM 256
#define BN 128
#define BK 32
#define LDP 40   // LDS row pitch in bf16 (80 B): 16B-aligned, conflict-free b128 reads

union FragBF16 { v16bf v; uint4 q[2]; };
union PackBF16 { __bf16 h[4]; uint2 u; };

__global__ __launch_bounds__(256)
void seg_linear_wmma_bf16(const float* __restrict__ x,      // [M][K]
                          const float* __restrict__ w,      // [N][K] (w_chunks flat)
                          const float* __restrict__ bias,   // [N]
                          const float* __restrict__ in_sc,  // [1]
                          const float* __restrict__ w_sc,   // [C]
                          float* __restrict__ out,          // [M][N]
                          int M, int N, int K)
{
    // double-buffered tiles: 40960 + 20480 = 61440 B of LDS
    __shared__ __bf16 sA[2][BM * LDP];
    __shared__ __bf16 sB[2][BN * LDP];

    const int tid  = threadIdx.x;
    const int lane = tid & 31;
    const int wave = tid >> 5;
    const int wm   = wave & 3;       // 0..3 : 64 rows of M each
    const int wn   = wave >> 2;      // 0..1 : 64 cols of N each
    const int lh   = lane >> 4;      // lane half (0/1)
    const int l16  = lane & 15;

    const int m0 = blockIdx.y * BM;
    const int n0 = blockIdx.x * BN;

    // hoist epilogue scalars so the tail doesn't wait on them
    const int   c  = n0 >> 10;       // whole 128-wide n-block lies in one 1024-chunk
    const float sc = in_sc[0] * w_sc[c];

    v8f acc[4][4];
#pragma unroll
    for (int mi = 0; mi < 4; ++mi)
#pragma unroll
        for (int ni = 0; ni < 4; ++ni) {
            v8f z = {0.f,0.f,0.f,0.f,0.f,0.f,0.f,0.f};
            acc[mi][ni] = z;
        }

    // ---- staging registers for one K-tile (issued early, stored late) ----
    float4 areg[8];   // A: 256 rows x 8 float4/row -> 8 per thread
    float4 breg[4];   // B: 128 rows x 8 float4/row -> 4 per thread

    auto load_global = [&](int kt) {
#pragma unroll
        for (int i = 0; i < 8; ++i) {
            const int f = tid + i * 256;
            areg[i] = *(const float4*)(x + (size_t)(m0 + (f >> 3)) * K + kt + ((f & 7) << 2));
        }
#pragma unroll
        for (int i = 0; i < 4; ++i) {
            const int f = tid + i * 256;
            breg[i] = *(const float4*)(w + (size_t)(n0 + (f >> 3)) * K + kt + ((f & 7) << 2));
        }
    };

    auto store_lds = [&](int buf) {
#pragma unroll
        for (int i = 0; i < 8; ++i) {
            const int f = tid + i * 256;
            PackBF16 p;
            p.h[0] = (__bf16)areg[i].x; p.h[1] = (__bf16)areg[i].y;
            p.h[2] = (__bf16)areg[i].z; p.h[3] = (__bf16)areg[i].w;
            *(uint2*)(&sA[buf][(f >> 3) * LDP + ((f & 7) << 2)]) = p.u;
        }
#pragma unroll
        for (int i = 0; i < 4; ++i) {
            const int f = tid + i * 256;
            PackBF16 p;
            p.h[0] = (__bf16)breg[i].x; p.h[1] = (__bf16)breg[i].y;
            p.h[2] = (__bf16)breg[i].z; p.h[3] = (__bf16)breg[i].w;
            *(uint2*)(&sB[buf][(f >> 3) * LDP + ((f & 7) << 2)]) = p.u;
        }
    };

    // A frag (16x32, M across lanes): lane<16 -> K{0..7,16..23}, lane>=16 -> K{8..15,24..31}
    auto load_afrag = [&](FragBF16& af, int buf, int mi) {
        const int row = wm * 64 + mi * 16 + l16;
        af.q[0] = *(const uint4*)(&sA[buf][row * LDP + 8 * lh]);
        af.q[1] = *(const uint4*)(&sA[buf][row * LDP + 16 + 8 * lh]);
    };

    auto compute = [&](int buf) {
        // B frags (32x16, N across lanes): lane<16 -> K{0..7,16..23}, lane>=16 -> K{8..15,24..31}
        FragBF16 bfrag[4];
#pragma unroll
        for (int ni = 0; ni < 4; ++ni) {
            const int row = wn * 64 + ni * 16 + l16;
            bfrag[ni].q[0] = *(const uint4*)(&sB[buf][row * LDP + 8 * lh]);
            bfrag[ni].q[1] = *(const uint4*)(&sB[buf][row * LDP + 16 + 8 * lh]);
        }
        // software-pipelined A fragments: issue group mi+1's ds_loads before
        // group mi's WMMAs so the dscnt wait skips the in-flight pair
        FragBF16 afr[2];
        load_afrag(afr[0], buf, 0);
#pragma unroll
        for (int mi = 0; mi < 4; ++mi) {
            if (mi < 3) load_afrag(afr[(mi + 1) & 1], buf, mi + 1);
#pragma unroll
            for (int ni = 0; ni < 4; ++ni)
                acc[mi][ni] = __builtin_amdgcn_wmma_f32_16x16x32_bf16(
                    false, afr[mi & 1].v, false, bfrag[ni].v,
                    (short)0, acc[mi][ni], false, false);
        }
    };

    // ---- pipelined main loop: 1 barrier per K-step, loads issued under WMMAs ----
    const int NK = K / BK;
    load_global(0);
    store_lds(0);
    __syncthreads();

    for (int kt = 0; kt < NK - 1; ++kt) {
        const int buf = kt & 1;
        load_global((kt + 1) * BK);  // global b128 loads in flight during WMMAs
        compute(buf);
        store_lds(buf ^ 1);          // cvt + ds_store after compute (waits on loads here)
        __syncthreads();
    }
    compute((NK - 1) & 1);

    // ---- epilogue: scale + bias, store f32 ----
#pragma unroll
    for (int ni = 0; ni < 4; ++ni) {
        const int   ncol = n0 + wn * 64 + ni * 16 + l16;
        const float bv   = bias[ncol];
#pragma unroll
        for (int mi = 0; mi < 4; ++mi) {
            const int mrow = m0 + wm * 64 + mi * 16 + lh * 8;  // D: vgpr r -> M = r + 8*lanehalf
            float* op = out + (size_t)mrow * N + ncol;
#pragma unroll
            for (int r = 0; r < 8; ++r)
                op[(size_t)r * N] = acc[mi][ni][r] * sc + bv;
        }
    }
}

extern "C" void kernel_launch(void* const* d_in, const int* in_sizes, int n_in,
                              void* d_out, int out_size, void* d_ws, size_t ws_size,
                              hipStream_t stream) {
    const float* x     = (const float*)d_in[0];
    const float* w     = (const float*)d_in[1];
    const float* bias  = (const float*)d_in[2];
    const float* in_sc = (const float*)d_in[3];
    const float* w_sc  = (const float*)d_in[4];
    float* out = (float*)d_out;

    const int K = 4096;
    const int N = in_sizes[2];            // bias length = C*Nc = 4096
    const int M = in_sizes[0] / K;        // 8192

    dim3 grid(N / BN, M / BM);            // x = n-block fastest: A panel reuse in L2
    seg_linear_wmma_bf16<<<grid, 256, 0, stream>>>(x, w, bias, in_sc, w_sc, out, M, N, K);
}
